// CrossAttentionModule_30296699306187
// MI455X (gfx1250) — matrix operand order repaired and checked by reading, not compiled, not run on hardware
//
#include <hip/hip_runtime.h>
#include <math.h>
#include <stdint.h>

typedef __attribute__((ext_vector_type(16))) _Float16 v16h;
typedef __attribute__((ext_vector_type(8)))  float    v8f;
typedef __attribute__((ext_vector_type(4)))  float    vf4;
typedef __attribute__((ext_vector_type(4)))  _Float16 vh4;
typedef __attribute__((ext_vector_type(8)))  _Float16 vh8;

#define HDIM   768
#define DHEAD  64
#define NHEADS 12
#define BATCH  4
#define LQ     1024
#define LKV    2048
#define PAD    8      // LDS rows 80B-strided: 16B-aligned b128 frag loads, conflict-free

__device__ __forceinline__ v8f wmma_f16(v16h a, v16h b, v8f c) {
  return __builtin_amdgcn_wmma_f32_16x16x32_f16(false, a, false, b, (short)0, c,
                                                false, false);
}

// CDNA5 async copy global->LDS (ASYNCcnt); per-lane LDS dest address in VGPR,
// per-lane 64-bit global source address in VGPR pair.
__device__ __forceinline__ void async_load_b128(unsigned lds_addr, const void* gaddr) {
  asm volatile("global_load_async_to_lds_b128 %0, %1, off"
               :: "v"(lds_addr), "v"((unsigned long long)(uintptr_t)gaddr)
               : "memory");
}
__device__ __forceinline__ void wait_async() {
  asm volatile("s_wait_asynccnt 0x0" ::: "memory");
}
// Low 32 bits of a flat shared-memory address are the LDS byte offset
// (ISA: aperture is identified by ADDR[63:32] only).
__device__ __forceinline__ unsigned lds_off(const void* p) {
  return (unsigned)(uintptr_t)p;
}

__device__ __forceinline__ vh4 to_h4(vf4 a) {
  vh4 r;
  r[0] = (_Float16)a[0]; r[1] = (_Float16)a[1];
  r[2] = (_Float16)a[2]; r[3] = (_Float16)a[3];
  return r;
}
__device__ __forceinline__ vh4 to_h4(vh4 a) { return a; }

template <typename T> struct V4;
template <> struct V4<float>    { typedef vf4 type; };
template <> struct V4<_Float16> { typedef vh4 type; };

// A fragment, 16x32 f16 row-major (leading dim ld).
__device__ __forceinline__ v16h load_a_frag(const _Float16* src, int ld, int lane) {
  int m = lane & 15, hi = lane >> 4;
  v16h a;
#pragma unroll
  for (int j = 0; j < 8; ++j) {
    int kk = ((j >> 2) << 4) + (hi << 3) + ((j & 3) << 1);
    a[2 * j]     = src[m * ld + kk];
    a[2 * j + 1] = src[m * ld + kk + 1];
  }
  return a;
}

// B fragment (32x16), element B[kk][n] from N-major storage: src[n*ld + kk]
__device__ __forceinline__ v16h load_bt_frag(const _Float16* src, int ld, int lane) {
  int n = lane & 15, hi = lane >> 4;
  v16h b;
#pragma unroll
  for (int j = 0; j < 8; ++j) {
    int kk = ((j >> 2) << 4) + (hi << 3) + ((j & 3) << 1);
    b[2 * j]     = src[n * ld + kk];
    b[2 * j + 1] = src[n * ld + kk + 1];
  }
  return b;
}

// B fragment (32x16), element B[kk][n] from K-major storage: src[kk*ld + n]
__device__ __forceinline__ v16h load_b_frag(const _Float16* src, int ld, int lane) {
  int n = lane & 15, hi = lane >> 4;
  v16h b;
#pragma unroll
  for (int j = 0; j < 8; ++j) {
    int kk = ((j >> 2) << 4) + (hi << 3) + ((j & 3) << 1);
    b[2 * j]     = src[kk * ld + n];
    b[2 * j + 1] = src[(kk + 1) * ld + n];
  }
  return b;
}

// ---------------------------------------------------------------------------
// GEMM: out[m][n] = sum_k X[m][k]*W[n][k] + bias[n] (+ res[m][n])
// 128 threads = 4 waves; 64x64 tile; K-step 32; double-buffered LDS with
// register staging (f32->f16 conversion requires the VGPR round trip).
// ---------------------------------------------------------------------------
template <typename InT, typename OutT, bool HASRES>
__global__ __launch_bounds__(128) void gemm_nt(
    const InT* __restrict__ X, const float* __restrict__ W,
    const float* __restrict__ bias, const float* __restrict__ res,
    OutT* __restrict__ out)
{
  __shared__ _Float16 As[2][64][32 + PAD];
  __shared__ _Float16 Ws[2][64][32 + PAD];

  const int tid  = threadIdx.x;
  const int lane = tid & 31;
  const int wave = tid >> 5;
  const int n0 = blockIdx.x * 64;
  const int m0 = blockIdx.y * 64;

  typedef typename V4<InT>::type InT4;
  InT4 ra[4];
  vf4  rw[4];

  auto stage_load = [&](int k0) {
#pragma unroll
    for (int i = 0; i < 4; ++i) {            // 8 independent vector loads in flight
      int e = (tid + i * 128) * 4;           // element index in 64x32 tile
      int r = e >> 5, c = e & 31;
      ra[i] = *(const InT4*)(X + (size_t)(m0 + r) * HDIM + k0 + c);
      rw[i] = *(const vf4*)(W + (size_t)(n0 + r) * HDIM + k0 + c);
    }
  };
  auto stage_store = [&](int buf) {
#pragma unroll
    for (int i = 0; i < 4; ++i) {
      int e = (tid + i * 128) * 4;
      int r = e >> 5, c = e & 31;
      *(vh4*)&As[buf][r][c] = to_h4(ra[i]);
      *(vh4*)&Ws[buf][r][c] = to_h4(rw[i]);
    }
  };

  const v8f vzero = {0.f, 0.f, 0.f, 0.f, 0.f, 0.f, 0.f, 0.f};
  v8f acc[4];
#pragma unroll
  for (int t = 0; t < 4; ++t) acc[t] = vzero;

  constexpr int KT = HDIM / 32;  // 24
  stage_load(0);
  stage_store(0);
  __syncthreads();

  for (int kt = 0; kt < KT; ++kt) {
    const int cur = kt & 1;
    if (kt + 1 < KT) stage_load((kt + 1) * 32);   // next tile's loads issue early

    v16h a = load_a_frag(&As[cur][wave * 16][0], 32 + PAD, lane);
#pragma unroll
    for (int t = 0; t < 4; ++t)
      acc[t] = wmma_f16(a, load_bt_frag(&Ws[cur][t * 16][0], 32 + PAD, lane), acc[t]);

    if (kt + 1 < KT) stage_store((kt + 1) & 1);
    __syncthreads();
  }

  const int nl = lane & 15, hi = lane >> 4;
#pragma unroll
  for (int t = 0; t < 4; ++t) {
#pragma unroll
    for (int r = 0; r < 8; ++r) {
      int gm = m0 + wave * 16 + hi * 8 + r;
      int gn = n0 + t * 16 + nl;
      float val = acc[t][r] + bias[gn];
      if (HASRES) val += res[(size_t)gm * HDIM + gn];
      out[(size_t)gm * HDIM + gn] = (OutT)val;
    }
  }
}

// ---------------------------------------------------------------------------
// Flash attention over f16 q/k/v: block = 64 q-rows of one (batch, head),
// 4 waves x 16 rows. Double-buffered 64-wide K/V tiles staged with the CDNA5
// async copy engine (global_load_async_to_lds_b128, ASYNCcnt) so the DMA
// overlaps WMMA + f32 online softmax on the current tile. Softmax scale (1/8)
// folded into the Q fragment.
// ---------------------------------------------------------------------------
__global__ __launch_bounds__(128) void attn_flash(
    const _Float16* __restrict__ q, const _Float16* __restrict__ k,
    const _Float16* __restrict__ v, const int* __restrict__ mask,
    _Float16* __restrict__ ctx)
{
  __shared__ _Float16 Kt[2][64][DHEAD + PAD];
  __shared__ _Float16 Vt[2][64][DHEAD + PAD];
  __shared__ _Float16 Pt[4][16][64 + PAD];

  const int tid  = threadIdx.x;
  const int lane = tid & 31;
  const int wave = tid >> 5;
  const int nl = lane & 15, hi = lane >> 4;

  const int bx = blockIdx.x;
  const int qt = bx & 15;                 // LQ/64 = 16 q-tiles
  const int h  = (bx >> 4) % NHEADS;
  const int b  = bx / (16 * NHEADS);

  const size_t qrow0 = (size_t)b * LQ + (size_t)qt * 64 + (size_t)wave * 16;
  const _Float16* qbase = q + qrow0 * HDIM + h * DHEAD;
  const _Float16* kb = k + ((size_t)b * LKV) * HDIM + h * DHEAD;
  const _Float16* vb = v + ((size_t)b * LKV) * HDIM + h * DHEAD;
  const int*      mb = mask + b * LKV;

  // Q fragments (2 K-chunks of 32), contiguous b128 loads per lane-half,
  // pre-scaled by 1/sqrt(64) = 0.125 (exact in f16).
  const _Float16 hs = (_Float16)0.125f;
  v16h aq[2];
#pragma unroll
  for (int c = 0; c < 2; ++c) {
    vh8 l0 = *(const vh8*)(qbase + (size_t)nl * HDIM + c * 32 + hi * 8);
    vh8 l1 = *(const vh8*)(qbase + (size_t)nl * HDIM + c * 32 + 16 + hi * 8);
#pragma unroll
    for (int u = 0; u < 8; ++u) {
      aq[c][u]     = l0[u] * hs;
      aq[c][8 + u] = l1[u] * hs;
    }
  }

  // Async-DMA one 64x64 f16 K tile + V tile into LDS buffer `buf`.
  auto kv_stage = [&](int kv0, int buf) {
#pragma unroll
    for (int i = 0; i < 4; ++i) {
      int e = tid + i * 128;                // 512 chunks of 8 halves per tile
      int r = e >> 3, c8 = (e & 7) * 8;
      async_load_b128(lds_off(&Kt[buf][r][c8]), kb + (size_t)(kv0 + r) * HDIM + c8);
      async_load_b128(lds_off(&Vt[buf][r][c8]), vb + (size_t)(kv0 + r) * HDIM + c8);
    }
  };

  const v8f vzero = {0.f, 0.f, 0.f, 0.f, 0.f, 0.f, 0.f, 0.f};
  float rowmax[8], rowsum[8];
  v8f o[4];
#pragma unroll
  for (int r = 0; r < 8; ++r) { rowmax[r] = -1e30f; rowsum[r] = 0.f; }
#pragma unroll
  for (int t = 0; t < 4; ++t) o[t] = vzero;

  constexpr int NKVT = LKV / 64;  // 32
  kv_stage(0, 0);
  wait_async();
  __syncthreads();

  for (int kt = 0; kt < NKVT; ++kt) {
    const int cur = kt & 1;
    const int kv0 = kt * 64;
    // Start the DMA for the next tile: buffer cur^1 was last read before the
    // barrier that ended the previous iteration, so it is free to overwrite.
    if (kt + 1 < NKVT) kv_stage((kt + 1) * 64, cur ^ 1);

    // S = (Q*scale) * K^T : 16 q-rows x 64 kv-cols per wave
    float st[4][8];
#pragma unroll
    for (int t = 0; t < 4; ++t) {
      v8f s = vzero;
#pragma unroll
      for (int c = 0; c < 2; ++c)
        s = wmma_f16(aq[c], load_bt_frag(&Kt[cur][t * 16][0] + c * 32, DHEAD + PAD, lane), s);
      float madd = (float)mb[kv0 + t * 16 + nl] * -10000.f;
#pragma unroll
      for (int r = 0; r < 8; ++r) st[t][r] = s[r] + madd;
    }

    // Online softmax: C-tile row = hi*8 + r lives in one 16-lane half
#pragma unroll
    for (int r = 0; r < 8; ++r) {
      float mx = fmaxf(fmaxf(st[0][r], st[1][r]), fmaxf(st[2][r], st[3][r]));
#pragma unroll
      for (int off = 1; off < 16; off <<= 1)
        mx = fmaxf(mx, __shfl_xor(mx, off, 32));
      float mnew  = fmaxf(rowmax[r], mx);
      float alpha = __expf(rowmax[r] - mnew);
      rowmax[r] = mnew;
      float ps = 0.f;
#pragma unroll
      for (int t = 0; t < 4; ++t) {
        float p = __expf(st[t][r] - mnew);
        st[t][r] = p;
        ps += p;
      }
#pragma unroll
      for (int off = 1; off < 16; off <<= 1)
        ps += __shfl_xor(ps, off, 32);
      rowsum[r] = rowsum[r] * alpha + ps;
#pragma unroll
      for (int t = 0; t < 4; ++t) o[t][r] *= alpha;
    }

    // Re-lay P into A-fragment form through LDS
#pragma unroll
    for (int t = 0; t < 4; ++t)
#pragma unroll
      for (int r = 0; r < 8; ++r)
        Pt[wave][hi * 8 + r][t * 16 + nl] = (_Float16)st[t][r];

    __syncthreads();

    // O += P * V
#pragma unroll
    for (int c = 0; c < 2; ++c) {
      v16h ap = load_a_frag(&Pt[wave][0][0] + c * 32, 64 + PAD, lane);
#pragma unroll
      for (int t = 0; t < 4; ++t)
        o[t] = wmma_f16(ap, load_b_frag(&Vt[cur][c * 32][t * 16], DHEAD + PAD, lane), o[t]);
    }

    // Next tile's DMA must be complete (and all waves done reading cur)
    // before the next iteration touches LDS.
    wait_async();
    __syncthreads();
  }

  // Normalize and write f16 ctx (flat [B*Lq, HDIM], head h at cols h*64..)
  _Float16* cbase = ctx + qrow0 * HDIM + h * DHEAD;
#pragma unroll
  for (int r = 0; r < 8; ++r) {
    float inv = 1.f / rowsum[r];
#pragma unroll
    for (int t = 0; t < 4; ++t)
      cbase[(size_t)(hi * 8 + r) * HDIM + t * 16 + nl] = (_Float16)(o[t][r] * inv);
  }
}

// ---------------------------------------------------------------------------
// LayerNorm over last dim (768); one block per row.
// ---------------------------------------------------------------------------
__global__ __launch_bounds__(256) void ln_kernel(
    const float* __restrict__ in, const float* __restrict__ gamma,
    const float* __restrict__ beta, float* __restrict__ out)
{
  __shared__ float rs[256];
  __shared__ float rs2[256];
  const int tid = threadIdx.x;
  const size_t row = blockIdx.x;
  const float* x = in + row * HDIM;

  float s = 0.f, s2 = 0.f;
  for (int i = tid; i < HDIM; i += 256) {
    float val = x[i];
    s += val;
    s2 += val * val;
  }
  rs[tid] = s;
  rs2[tid] = s2;
  __syncthreads();
  for (int o = 128; o > 0; o >>= 1) {
    if (tid < o) { rs[tid] += rs[tid + o]; rs2[tid] += rs2[tid + o]; }
    __syncthreads();
  }
  const float mu  = rs[0] * (1.f / HDIM);
  const float var = rs2[0] * (1.f / HDIM) - mu * mu;
  const float inv = rsqrtf(var + 1e-5f);
  for (int i = tid; i < HDIM; i += 256)
    out[row * HDIM + i] = (x[i] - mu) * inv * gamma[i] + beta[i];
}

// ---------------------------------------------------------------------------
extern "C" void kernel_launch(void* const* d_in, const int* in_sizes, int n_in,
                              void* d_out, int out_size, void* d_ws, size_t ws_size,
                              hipStream_t stream) {
  (void)in_sizes; (void)n_in; (void)out_size; (void)ws_size;

  const float* x     = (const float*)d_in[0];   // query_states [4,1024,768]
  const float* kvx   = (const float*)d_in[1];   // key_value_states [4,2048,768]
  const int*   msk   = (const int*)d_in[2];     // [4,2048]
  const float* Wq    = (const float*)d_in[3];
  const float* bq    = (const float*)d_in[4];
  const float* Wk    = (const float*)d_in[5];
  const float* bk    = (const float*)d_in[6];
  const float* Wv    = (const float*)d_in[7];
  const float* bv    = (const float*)d_in[8];
  const float* Wo    = (const float*)d_in[9];
  const float* bo    = (const float*)d_in[10];
  const float* gamma = (const float*)d_in[11];
  const float* beta  = (const float*)d_in[12];
  float* outp = (float*)d_out;

  const size_t NQ  = (size_t)BATCH * LQ;    // 4096 rows
  const size_t NKV = (size_t)BATCH * LKV;   // 8192 rows
  _Float16* qb = (_Float16*)d_ws;           // [NQ,  768] f16
  _Float16* kb = qb + NQ * HDIM;            // [NKV, 768] f16
  _Float16* vb = kb + NKV * HDIM;           // [NKV, 768] f16
  _Float16* cb = vb + NKV * HDIM;           // [NQ,  768] f16 attention context
  float*    pb = (float*)(cb + NQ * HDIM);  // [NQ,  768] f32 out-proj + residual

  dim3 blk(128);
  // Q/K/V projections (f32 in, f16 out)
  gemm_nt<float, _Float16, false>
      <<<dim3(HDIM / 64, NQ / 64), blk, 0, stream>>>(x, Wq, bq, nullptr, qb);
  gemm_nt<float, _Float16, false>
      <<<dim3(HDIM / 64, NKV / 64), blk, 0, stream>>>(kvx, Wk, bk, nullptr, kb);
  gemm_nt<float, _Float16, false>
      <<<dim3(HDIM / 64, NKV / 64), blk, 0, stream>>>(kvx, Wv, bv, nullptr, vb);
  // Flash attention: one block per (batch, head, 64-row q-tile)
  attn_flash<<<dim3(BATCH * NHEADS * (LQ / 64)), blk, 0, stream>>>(qb, kb, vb, msk, cb);
  // Output projection + residual (f16 in, f32 out)
  gemm_nt<_Float16, float, true>
      <<<dim3(HDIM / 64, NQ / 64), blk, 0, stream>>>(cb, Wo, bo, x, pb);
  // LayerNorm
  ln_kernel<<<dim3((unsigned)NQ), dim3(256), 0, stream>>>(pb, gamma, beta, outp);
}
